// ComputeREC_39608188404007
// MI455X (gfx1250) — matrix operand order repaired and checked by reading, not compiled, not run on hardware
//
#include <hip/hip_runtime.h>
#include <hip/hip_bf16.h>
#include <math.h>

// Problem sizes (from reference)
#define IN_F   512
#define HID    512
#define NCLS   512
#define BATCH  32
#define WW     2048
#define ROWS   16          // W-rows per workgroup
#define THREADS 256        // 8 waves of 32
#define K1     1024        // GEMM1 K = IN_F + HID

typedef __attribute__((ext_vector_type(16))) __bf16 v16bf;
typedef __attribute__((ext_vector_type(8)))  float  v8f;

union Frag16 { int4 i[2]; v16bf v; unsigned short u[16]; };

__device__ __forceinline__ unsigned short f2bf(float f) {
  unsigned int u = __float_as_uint(f);
  u += 0x7FFFu + ((u >> 16) & 1u);           // round-to-nearest-even
  return (unsigned short)(u >> 16);
}
__device__ __forceinline__ float bf2f(unsigned short h) {
  return __uint_as_float(((unsigned int)h) << 16);
}

// Branch-free fast tanh: 1 - 2/(exp(2x)+1).
// Saturates correctly for large |x| (exp -> inf/0), no EXEC divergence.
__device__ __forceinline__ float fast_tanh(float x) {
  float e = __expf(2.0f * x);                      // v_exp_f32
  return 1.0f - 2.0f * __builtin_amdgcn_rcpf(e + 1.0f);  // v_rcp_f32
}

// Pre-swizzle an f32 weight matrix (K x 512, row-major) into bf16 WMMA
// B-fragment order: dst[((kt*32 + ntile)*32 + lane)*16 + e]
// where e in [0,16): K = kt*32 + (lane>>4)*16 + e, N = ntile*16 + (lane&15).
__global__ __launch_bounds__(THREADS) void swizzle_w(const float* __restrict__ src,
                                                     unsigned short* __restrict__ dst,
                                                     int K) {
  int id   = blockIdx.x * THREADS + threadIdx.x;
  int lane = id & 31;
  int tile = id >> 5;                 // kt * 32 + ntile  (32 n-tiles of 16)
  int kt   = tile >> 5;
  if (kt >= (K >> 5)) return;
  int nt   = tile & 31;
  int n    = nt * 16 + (lane & 15);
  int kb   = kt * 32 + ((lane >> 4) << 4);
  Frag16 f;
  #pragma unroll
  for (int e = 0; e < 16; ++e) f.u[e] = f2bf(src[(size_t)(kb + e) * 512 + n]);
  int4* p = (int4*)(dst + ((size_t)tile * 32 + lane) * 16);
  p[0] = f.i[0];
  p[1] = f.i[1];
}

__global__ __launch_bounds__(THREADS) void rnn_step_kernel(
    const float* __restrict__ input, const float* __restrict__ pre_state,
    const float* __restrict__ b1, const float* __restrict__ b2,
    const unsigned short* __restrict__ W1s, const unsigned short* __restrict__ W2s,
    float* __restrict__ out, float* __restrict__ final_state)
{
  // sA[row][k]: k in [0,512) = x_t (bf16), k in [512,1024) = state/hidden (bf16)
  __shared__ unsigned short sA[ROWS * 1024];   // 32 KB
  __shared__ float          sL[ROWS * 512];    // 32 KB logits staging

  const int tid  = threadIdx.x;
  const int lane = tid & 31;
  const int wave = tid >> 5;                   // 0..7, owns cols [wave*64, wave*64+64)
  const int row0 = blockIdx.x * ROWS;
  const int ln   = lane & 15;                  // A: M row; B/C/D: N col
  const int g    = lane >> 4;                  // lane half-group

  // Per-wave bias fragments (t-invariant): C[m][n] = b[n] broadcast over m.
  float bias1[4], bias2[4];
  #pragma unroll
  for (int nt = 0; nt < 4; ++nt) {
    bias1[nt] = b1[wave * 64 + nt * 16 + ln];
    bias2[nt] = b2[wave * 64 + nt * 16 + ln];
  }

  // Initialize recurrent state from pre_state (f32 -> bf16).
  for (int i = tid; i < ROWS * 512; i += THREADS) {
    int r = i >> 9, c = i & 511;
    sA[r * 1024 + 512 + c] = f2bf(pre_state[(size_t)(row0 + r) * 512 + c]);
  }

  #pragma unroll 1
  for (int t = 0; t < BATCH; ++t) {
    __syncthreads();  // prior-step consumers of sA (x half) / sL are done

    // Load x_t tile, f32 -> bf16, vectorized (float4 in, 8B packed LDS store).
    #pragma unroll 1
    for (int i = tid; i < ROWS * 128; i += THREADS) {
      int r = i >> 7, c4 = i & 127;
      const float4 v = *(const float4*)(input + ((size_t)t * WW + row0 + r) * 512 + c4 * 4);
      union { unsigned short u[4]; unsigned long long q; } pk;
      pk.u[0] = f2bf(v.x); pk.u[1] = f2bf(v.y); pk.u[2] = f2bf(v.z); pk.u[3] = f2bf(v.w);
      *(unsigned long long*)(&sA[r * 1024 + c4 * 4]) = pk.q;
    }
    __syncthreads();

    // ---- GEMM1: hidden = tanh([x|state] @ W1 + b1); K = 1024 ----
    v8f acc[4];
    #pragma unroll
    for (int nt = 0; nt < 4; ++nt) {
      #pragma unroll
      for (int r = 0; r < 8; ++r) acc[nt][r] = bias1[nt];
    }
    #pragma unroll 1
    for (int kt = 0; kt < 32; ++kt) {
      Frag16 a;  // A 16x32 bf16: lane ln holds K kb+g*16..+15
      const unsigned short* ap = &sA[ln * 1024 + kt * 32 + g * 16];
      a.i[0] = *(const int4*)(ap);
      a.i[1] = *(const int4*)(ap + 8);
      #pragma unroll
      for (int nt = 0; nt < 4; ++nt) {
        Frag16 b;
        const unsigned short* bp =
            W1s + (((size_t)kt * 32 + (wave * 4 + nt)) * 32 + lane) * 16;
        b.i[0] = ((const int4*)bp)[0];
        b.i[1] = ((const int4*)bp)[1];
        acc[nt] = __builtin_amdgcn_wmma_f32_16x16x32_bf16(
            false, a.v, false, b.v, (short)0, acc[nt], false, false);
      }
    }
    __syncthreads();  // everyone finished reading the OLD state

    // tanh, write hidden (bf16) into the state half of sA.
    #pragma unroll
    for (int nt = 0; nt < 4; ++nt) {
      const int col = wave * 64 + nt * 16 + ln;
      #pragma unroll
      for (int r = 0; r < 8; ++r) {
        float h = fast_tanh(acc[nt][r]);
        sA[(r + 8 * g) * 1024 + 512 + col] = f2bf(h);
      }
    }
    __syncthreads();

    // ---- GEMM2: logits = hidden @ W2 + b2; K = 512 ----
    #pragma unroll
    for (int nt = 0; nt < 4; ++nt) {
      #pragma unroll
      for (int r = 0; r < 8; ++r) acc[nt][r] = bias2[nt];
    }
    #pragma unroll 1
    for (int kt = 0; kt < 16; ++kt) {
      Frag16 a;
      const unsigned short* ap = &sA[ln * 1024 + 512 + kt * 32 + g * 16];
      a.i[0] = *(const int4*)(ap);
      a.i[1] = *(const int4*)(ap + 8);
      #pragma unroll
      for (int nt = 0; nt < 4; ++nt) {
        Frag16 b;
        const unsigned short* bp =
            W2s + (((size_t)kt * 32 + (wave * 4 + nt)) * 32 + lane) * 16;
        b.i[0] = ((const int4*)bp)[0];
        b.i[1] = ((const int4*)bp)[1];
        acc[nt] = __builtin_amdgcn_wmma_f32_16x16x32_bf16(
            false, a.v, false, b.v, (short)0, acc[nt], false, false);
      }
    }
    // Stage logits to LDS for the cross-column softmax reduction.
    #pragma unroll
    for (int nt = 0; nt < 4; ++nt) {
      const int col = wave * 64 + nt * 16 + ln;
      #pragma unroll
      for (int r = 0; r < 8; ++r)
        sL[(r + 8 * g) * 512 + col] = acc[nt][r];
    }
    __syncthreads();

    // ---- Softmax over 512 cols; each wave reduces rows {2w, 2w+1} ----
    #pragma unroll
    for (int rr = 0; rr < 2; ++rr) {
      const int row = wave * 2 + rr;
      float vals[16];
      float vmax = -INFINITY;
      #pragma unroll
      for (int j = 0; j < 16; ++j) {
        vals[j] = sL[row * 512 + j * 32 + lane];
        vmax = fmaxf(vmax, vals[j]);
      }
      #pragma unroll
      for (int off = 16; off > 0; off >>= 1)
        vmax = fmaxf(vmax, __shfl_xor(vmax, off, 32));
      float sum = 0.f;
      #pragma unroll
      for (int j = 0; j < 16; ++j) { vals[j] = __expf(vals[j] - vmax); sum += vals[j]; }
      #pragma unroll
      for (int off = 16; off > 0; off >>= 1)
        sum += __shfl_xor(sum, off, 32);
      const float inv = __builtin_amdgcn_rcpf(sum);   // v_rcp_f32
      float* op = out + ((size_t)t * WW + row0 + row) * 512;
      #pragma unroll
      for (int j = 0; j < 16; ++j) op[j * 32 + lane] = vals[j] * inv;
    }
  }

  __syncthreads();
  // Emit final state (bf16 -> f32).
  for (int i = tid; i < ROWS * 512; i += THREADS) {
    int r = i >> 9, c = i & 511;
    final_state[(size_t)(row0 + r) * 512 + c] = bf2f(sA[r * 1024 + 512 + c]);
  }
}

extern "C" void kernel_launch(void* const* d_in, const int* in_sizes, int n_in,
                              void* d_out, int out_size, void* d_ws, size_t ws_size,
                              hipStream_t stream) {
  const float* input     = (const float*)d_in[0];
  const float* pre_state = (const float*)d_in[1];
  const float* W1        = (const float*)d_in[2];
  const float* b1        = (const float*)d_in[3];
  const float* W2        = (const float*)d_in[4];
  const float* b2        = (const float*)d_in[5];

  float* out         = (float*)d_out;
  float* final_state = out + (size_t)BATCH * WW * NCLS;

  unsigned short* W1s = (unsigned short*)d_ws;            // 1024*512 bf16 = 1 MB
  unsigned short* W2s = W1s + (size_t)K1 * HID;           //  512*512 bf16 = 512 KB

  // Pre-convert + swizzle weights into WMMA B-fragment order (bf16).
  swizzle_w<<<(32 * 32 * 32) / THREADS, THREADS, 0, stream>>>(W1, W1s, 1024);
  swizzle_w<<<(16 * 32 * 32) / THREADS, THREADS, 0, stream>>>(W2, W2s, 512);

  // Recurrent kernel: 128 workgroups x 16 rows of W, 32 sequential steps.
  rnn_step_kernel<<<WW / ROWS, THREADS, 0, stream>>>(
      input, pre_state, b1, b2, W1s, W2s, out, final_state);
}
